// Model_39676907885601
// MI455X (gfx1250) — matrix operand order repaired and checked by reading, not compiled, not run on hardware
//
#include <hip/hip_runtime.h>
#include <hip/hip_bf16.h>

typedef __attribute__((ext_vector_type(16))) __bf16 v16bf;
typedef __attribute__((ext_vector_type(8)))  __bf16 v8bf;
typedef __attribute__((ext_vector_type(4)))  __bf16 v4bf;
typedef __attribute__((ext_vector_type(2)))  __bf16 v2bf;
typedef __attribute__((ext_vector_type(8)))  float  v8f;
typedef __attribute__((ext_vector_type(4)))  float  v4f;
typedef __attribute__((ext_vector_type(4)))  int    v4i;
typedef __attribute__((ext_vector_type(8)))  int    v8i;

#define BATCH   16
#define SEQ     2048
#define DIM     64
#define QTILE   16
#define KTILE   32
#define WAVES   8
#define QBLOCK  (WAVES * QTILE)      // 128 queries per block
// score scale folded into exp2 domain: 0.125 * log2(e)
#define SLOG2E  0.18033688011112042f

// padded LDS strides (elements); stride*2 bytes multiple of 16 for b128 ops
#define KSTR    72                   // [key][d]  row stride
#define TSTR    40                   // [d][key]  row stride

static __device__ inline v16bf cat16(v8bf lo, v8bf hi) {
    return __builtin_shufflevector(lo, hi, 0,1,2,3,4,5,6,7,8,9,10,11,12,13,14,15);
}

__global__ __launch_bounds__(256)
void fa_fwd_kernel(const float* __restrict__ x1, const float* __restrict__ x2,
                   float* __restrict__ out) {
    // double-buffered K/V tile in two layouts
    __shared__ __bf16 ldsK [2][KTILE * KSTR];   // [key][d]
    __shared__ __bf16 ldsKT[2][DIM * TSTR];     // [d][key]

    const int tid  = threadIdx.x;
    const int wave = tid >> 5;
    const int lane = tid & 31;
    const int half = lane >> 4;      // 0: lanes 0-15, 1: lanes 16-31
    const int l16  = lane & 15;

    const int b   = blockIdx.x >> 4;
    const int qbk = blockIdx.x & 15;
    const int q0  = qbk * QBLOCK + wave * QTILE;

    const float* qptr = x1 + (size_t)b * SEQ * DIM;
    const float* kptr = x2 + (size_t)b * SEQ * DIM;

    // ---- Q as B-fragments (d x query): N = query = l16, K = d = 32c + {8*half+e (e<8), 16+8*half+e-8 (e>=8)} ----
    v16bf qfB[2];
    {
        const float* qrow = qptr + (size_t)(q0 + l16) * DIM;
        #pragma unroll
        for (int c = 0; c < 2; ++c) {
            const float* p = qrow + 32 * c + 8 * half;
            v4f f0 = *(const v4f*)(p);
            v4f f1 = *(const v4f*)(p + 4);
            v4f f2 = *(const v4f*)(p + 16);
            v4f f3 = *(const v4f*)(p + 20);
            #pragma unroll
            for (int k = 0; k < 4; ++k) {
                qfB[c][k]      = (__bf16)f0[k];
                qfB[c][4 + k]  = (__bf16)f1[k];
                qfB[c][8 + k]  = (__bf16)f2[k];
                qfB[c][12 + k] = (__bf16)f3[k];
            }
        }
    }

    // O^T accumulators: tile t -> d = 16t + 8*half + i, query = l16
    v8f acc[4];
    #pragma unroll
    for (int t = 0; t < 4; ++t)
        #pragma unroll
        for (int i = 0; i < 8; ++i) acc[t][i] = 0.0f;
    float m = -3.0e38f, l = 0.0f;    // per-lane scalars (query = l16)

    // cooperative-load assignment: 2 keys x 4 d per thread
    const int kp = (tid >> 4) * 2;       // key pair base: 0,2,..,30
    const int d0 = (tid & 15) * 4;       // d start: 0..60

    // ---- stage tile 0 into buffer 0 ----
    {
        const float* src = kptr + (size_t)kp * DIM + d0;
        v4f a0 = *(const v4f*)(src);
        v4f a1 = *(const v4f*)(src + DIM);
        v4bf r0, r1;
        #pragma unroll
        for (int e = 0; e < 4; ++e) { r0[e] = (__bf16)a0[e]; r1[e] = (__bf16)a1[e]; }
        *(v4bf*)(&ldsK[0][kp * KSTR + d0])       = r0;   // b64
        *(v4bf*)(&ldsK[0][(kp + 1) * KSTR + d0]) = r1;   // b64
        #pragma unroll
        for (int e = 0; e < 4; ++e) {                    // packed key-pair b32
            v2bf pr; pr[0] = r0[e]; pr[1] = r1[e];
            *(v2bf*)(&ldsKT[0][(d0 + e) * TSTR + kp]) = pr;
        }
    }
    __syncthreads();

    int p = 0;
    for (int j = 0; j < SEQ; j += KTILE) {
        const bool has_next = (j + KTILE) < SEQ;

        // issue next tile's global loads early (latency hidden behind compute)
        v4f na0, na1;
        if (has_next) {
            const float* src = kptr + (size_t)(j + KTILE + kp) * DIM + d0;
            na0 = *(const v4f*)(src);
            na1 = *(const v4f*)(src + DIM);
            if (j + 2 * KTILE < SEQ)
                __builtin_prefetch(kptr + (size_t)(j + 2 * KTILE) * DIM, 0, 1);
        }

        const __bf16* Kb = &ldsK[p][0];
        const __bf16* Vb = &ldsKT[p][0];

        // ---- hoist ALL fragment loads: 16 ds_load_b128 into distinct regs so the
        //      scheduler can wait incrementally instead of s_wait_dscnt 0 per WMMA ----
        // K A-frags (16x32): M = key, K = d; elems {8h..8h+7, 16+8h..16+8h+7}
        const int dof0 = 8 * half;          // c = 0
        const int dof1 = 32 + 8 * half;     // c = 1
        v16bf kA00 = cat16(*(const v8bf*)(&Kb[(l16)      * KSTR + dof0]),
                           *(const v8bf*)(&Kb[(l16)      * KSTR + dof0 + 16]));
        v16bf kA10 = cat16(*(const v8bf*)(&Kb[(16 + l16) * KSTR + dof0]),
                           *(const v8bf*)(&Kb[(16 + l16) * KSTR + dof0 + 16]));
        v16bf kA01 = cat16(*(const v8bf*)(&Kb[(l16)      * KSTR + dof1]),
                           *(const v8bf*)(&Kb[(l16)      * KSTR + dof1 + 16]));
        v16bf kA11 = cat16(*(const v8bf*)(&Kb[(16 + l16) * KSTR + dof1]),
                           *(const v8bf*)(&Kb[(16 + l16) * KSTR + dof1 + 16]));
        // V^T A-frags from [d][key] copy: independent of scores -> latency hides
        // under the softmax VALU work below
        v16bf vA[4];
        #pragma unroll
        for (int t = 0; t < 4; ++t) {
            const __bf16* vrow = &Vb[(16 * t + l16) * TSTR + 16 * half];
            vA[t] = cat16(*(const v8bf*)(vrow), *(const v8bf*)(vrow + 8));
        }

        // ---- S^T = K * Q^T : C tiles are (key x query) ----
        v8f st0, st1;
        #pragma unroll
        for (int i = 0; i < 8; ++i) { st0[i] = 0.0f; st1[i] = 0.0f; }
        st0 = __builtin_amdgcn_wmma_f32_16x16x32_bf16(false, kA00, false, qfB[0],
                                                      (short)0, st0, false, false);
        st1 = __builtin_amdgcn_wmma_f32_16x16x32_bf16(false, kA10, false, qfB[0],
                                                      (short)0, st1, false, false);
        st0 = __builtin_amdgcn_wmma_f32_16x16x32_bf16(false, kA01, false, qfB[1],
                                                      (short)0, st0, false, false);
        st1 = __builtin_amdgcn_wmma_f32_16x16x32_bf16(false, kA11, false, qfB[1],
                                                      (short)0, st1, false, false);

        // ---- online softmax, exp2 domain; each lane owns query l16 ----
        #pragma unroll
        for (int i = 0; i < 8; ++i) { st0[i] *= SLOG2E; st1[i] *= SLOG2E; }

        float vmax = fmaxf(st0[0], st1[0]);
        #pragma unroll
        for (int i = 1; i < 8; ++i) vmax = fmaxf(vmax, fmaxf(st0[i], st1[i]));
        vmax = fmaxf(vmax, __shfl_xor(vmax, 16, 32));        // combine halves
        const float mn = fmaxf(m, vmax);
        const float alpha = __builtin_amdgcn_exp2f(m - mn);
        m = mn;

        #pragma unroll
        for (int i = 0; i < 8; ++i) {
            st0[i] = __builtin_amdgcn_exp2f(st0[i] - m);
            st1[i] = __builtin_amdgcn_exp2f(st1[i] - m);
        }
        float rs = st0[0] + st1[0];
        #pragma unroll
        for (int i = 1; i < 8; ++i) rs += st0[i] + st1[i];
        rs += __shfl_xor(rs, 16, 32);
        l = l * alpha + rs;

        #pragma unroll
        for (int t = 0; t < 4; ++t)
            #pragma unroll
            for (int i = 0; i < 8; ++i) acc[t][i] *= alpha;

        // ---- build P^T B-fragment: pack to bf16 dwords, one cross-half exchange ----
        // half0 keeps keys 0..7 (st0), needs partner's st0 (keys 8..15)
        // half1 keeps keys 24..31 (st1), needs partner's st1 (keys 16..23)
        v8bf p0b, p1b;
        #pragma unroll
        for (int i = 0; i < 8; ++i) { p0b[i] = (__bf16)st0[i]; p1b[i] = (__bf16)st1[i]; }
        const v4i s0d = __builtin_bit_cast(v4i, p0b);
        const v4i s1d = __builtin_bit_cast(v4i, p1b);
        v4i kd, sd, rd;
        #pragma unroll
        for (int jj = 0; jj < 4; ++jj) {
            kd[jj] = half ? s1d[jj] : s0d[jj];
            sd[jj] = half ? s0d[jj] : s1d[jj];
        }
        #pragma unroll
        for (int jj = 0; jj < 4; ++jj) rd[jj] = __shfl_xor(sd[jj], 16, 32);
        v8i fd;
        #pragma unroll
        for (int jj = 0; jj < 4; ++jj) {
            fd[jj]     = half ? rd[jj] : kd[jj];   // elems 0..7  (K = 16h + 0..7)
            fd[4 + jj] = half ? kd[jj] : rd[jj];   // elems 8..15 (K = 16h + 8..15)
        }
        const v16bf pfB = __builtin_bit_cast(v16bf, fd);

        // ---- O^T += V^T * P^T ----
        #pragma unroll
        for (int t = 0; t < 4; ++t)
            acc[t] = __builtin_amdgcn_wmma_f32_16x16x32_bf16(false, vA[t], false, pfB,
                                                             (short)0, acc[t], false, false);

        // ---- convert + stage next tile into the other buffer ----
        if (has_next) {
            v4bf r0, r1;
            #pragma unroll
            for (int e = 0; e < 4; ++e) { r0[e] = (__bf16)na0[e]; r1[e] = (__bf16)na1[e]; }
            __bf16* Kn = &ldsK[p ^ 1][0];
            __bf16* Vn = &ldsKT[p ^ 1][0];
            *(v4bf*)(&Kn[kp * KSTR + d0])       = r0;
            *(v4bf*)(&Kn[(kp + 1) * KSTR + d0]) = r1;
            #pragma unroll
            for (int e = 0; e < 4; ++e) {
                v2bf pr; pr[0] = r0[e]; pr[1] = r1[e];
                *(v2bf*)(&Vn[(d0 + e) * TSTR + kp]) = pr;
            }
        }
        __syncthreads();
        p ^= 1;
    }

    // ---- epilogue: per-lane scalar 1/l, vectorized b128 stores ----
    const float inv = 1.0f / l;
    float* orow = out + ((size_t)b * SEQ + q0 + l16) * DIM;
    #pragma unroll
    for (int t = 0; t < 4; ++t) {
        v4f o0, o1;
        #pragma unroll
        for (int i = 0; i < 4; ++i) {
            o0[i] = acc[t][i] * inv;
            o1[i] = acc[t][4 + i] * inv;
        }
        *(v4f*)(orow + 16 * t + 8 * half)     = o0;
        *(v4f*)(orow + 16 * t + 8 * half + 4) = o1;
    }
}

extern "C" void kernel_launch(void* const* d_in, const int* in_sizes, int n_in,
                              void* d_out, int out_size, void* d_ws, size_t ws_size,
                              hipStream_t stream) {
    (void)in_sizes; (void)n_in; (void)out_size; (void)d_ws; (void)ws_size;
    const float* x1 = (const float*)d_in[0];
    const float* x2 = (const float*)d_in[1];
    float* out = (float*)d_out;
    dim3 grid(BATCH * (SEQ / QBLOCK));   // 256 blocks
    dim3 block(WAVES * 32);              // 8 waves
    fa_fwd_kernel<<<grid, block, 0, stream>>>(x1, x2, out);
}
